// BitNetBlock_30356828848661
// MI455X (gfx1250) — matrix operand order repaired and checked by reading, not compiled, not run on hardware
//
#include <hip/hip_runtime.h>
#include <hip/hip_bf16.h>
#include <math.h>

#define S_LEN 2048
#define HDIM  2560
#define INTER 6912
#define KVDIM 640
#define NHEADS 20
#define NKV    5
#define DHEAD  128

typedef __bf16 bh;
typedef __attribute__((ext_vector_type(8)))  __bf16 v8bf;
typedef __attribute__((ext_vector_type(16))) __bf16 v16bf;
typedef __attribute__((ext_vector_type(8)))  float  v8f;
typedef __attribute__((ext_vector_type(4)))  int    v4i;

// ---------- CDNA5 async Global->LDS path (ASYNCcnt), guarded so compile can't regress
#if defined(__gfx1250__) && __has_builtin(__builtin_amdgcn_global_load_async_to_lds_b128)
#define HAVE_ASYNC_LDS 1
#else
#define HAVE_ASYNC_LDS 0
#endif

// copy 16B (8 x bf16) from global to LDS via GLOBAL_LOAD_ASYNC_TO_LDS_B128
static __device__ __forceinline__ void async_copy16(bh* dst, const bh* src) {
#if HAVE_ASYNC_LDS
  v4i* sg = (v4i*)src;  // strip const, reinterpret as int4 (generic)
  v4i* dl = (v4i*)dst;
  __builtin_amdgcn_global_load_async_to_lds_b128(
      (__attribute__((address_space(1))) v4i*)sg,
      (__attribute__((address_space(3))) v4i*)dl, 0, 0);
#else
  *(v8bf*)dst = *(const v8bf*)src;
#endif
}

static __device__ __forceinline__ void async_wait_all() {
#if HAVE_ASYNC_LDS
#if __has_builtin(__builtin_amdgcn_s_wait_asynccnt)
  __builtin_amdgcn_s_wait_asynccnt(0);
#else
  asm volatile("s_wait_asynccnt 0x0" ::: "memory");
#endif
#endif
}

// Build a 16-element bf16 A/B fragment from two contiguous 8-element (16B) chunks.
// Callers pass p1 = p0 + 8: lane half h holds K=16h..16h+15 (contiguous order, same for A and B).
static __device__ __forceinline__ v16bf frag2(const bh* p0, const bh* p1) {
  v8bf a = *(const v8bf*)p0;
  v8bf b = *(const v8bf*)p1;
  v16bf r;
#pragma unroll
  for (int i = 0; i < 8; ++i) { r[i] = a[i]; r[i + 8] = b[i]; }
  return r;
}

// ---------------- ternarize: w_t = sign(w)*(|w| > 0.7*absmean), scale = absmean ----
__global__ void ternarize_k(const float* __restrict__ w, bh* __restrict__ t,
                            float* __restrict__ sc, int K) {
  const int row = blockIdx.x;
  const int tid = threadIdx.x;
  const float* wr = w + (size_t)row * K;
  float s = 0.f;
  for (int i = tid; i < K; i += 256) s += fabsf(wr[i]);
  __shared__ float red[256];
  red[tid] = s; __syncthreads();
  for (int st = 128; st > 0; st >>= 1) {
    if (tid < st) red[tid] += red[tid + st];
    __syncthreads();
  }
  const float absmean = red[0] / (float)K;
  const float thr = 0.7f * absmean;
  bh* tr = t + (size_t)row * K;
  for (int i = tid; i < K; i += 256) {
    const float v = wr[i];
    const float q = (fabsf(v) > thr) ? (v > 0.f ? 1.f : -1.f) : 0.f;
    tr[i] = (bh)q;
  }
  if (tid == 0) sc[row] = absmean;
}

// ---------------- rmsnorm (f32 -> bf16) ----------------
__global__ void rmsnorm_k(const float* __restrict__ x, const float* __restrict__ g,
                          bh* __restrict__ out) {
  const int row = blockIdx.x, tid = threadIdx.x;
  const float* xr = x + (size_t)row * HDIM;
  float s = 0.f;
  for (int i = tid; i < HDIM; i += 256) { const float v = xr[i]; s += v * v; }
  __shared__ float red[256];
  red[tid] = s; __syncthreads();
  for (int st = 128; st > 0; st >>= 1) {
    if (tid < st) red[tid] += red[tid + st];
    __syncthreads();
  }
  const float r = rsqrtf(red[0] / (float)HDIM + 1e-5f);
  bh* orow = out + (size_t)row * HDIM;
  for (int i = tid; i < HDIM; i += 256) orow[i] = (bh)(xr[i] * r * g[i]);
}

// ---------------- tiled bf16 WMMA GEMM: out[M,N] = A[M,K] * B[N,K]^T * scale[N] (+resid)
// block = 256 threads = 8 waves (4x2), tile 128(M) x 128(N) x 32(K), double-buffered LDS,
// async Global->LDS staging overlapped with WMMA; ONE barrier per K-step.
template<bool RESID>
__global__ __launch_bounds__(256)
void gemm_tern(const bh* __restrict__ A, const bh* __restrict__ B,
               const float* __restrict__ scale, const float* __restrict__ resid,
               float* __restrict__ out, int N, int K) {
  __shared__ bh As[2][128][32];
  __shared__ bh Bs[2][128][32];
  const int tid  = threadIdx.x;
  const int wave = tid >> 5, lane = tid & 31;
  const int wm = wave >> 1, wn = wave & 1;        // 4 x 2 wave grid
  const int l16 = lane & 15, lhi = lane >> 4;
  const int mBlk = blockIdx.y * 128, nBlk = blockIdx.x * 128;

  v8f acc[2][4];
#pragma unroll
  for (int mt = 0; mt < 2; ++mt)
#pragma unroll
    for (int nt = 0; nt < 4; ++nt)
#pragma unroll
      for (int i = 0; i < 8; ++i) acc[mt][nt][i] = 0.f;

  const int lrow = tid >> 1, lcol = (tid & 1) * 16;  // 128 rows x 32 cols, 16 bf16/thread
  const bh* aRow = A + (size_t)(mBlk + lrow) * K + lcol;
  const bh* bRow = B + (size_t)(nBlk + lrow) * K + lcol;

  auto stage = [&](int buf, int kt) {
    const bh* ap = aRow + kt * 32;
    const bh* bp = bRow + kt * 32;
    async_copy16(&As[buf][lrow][lcol],     ap);
    async_copy16(&As[buf][lrow][lcol + 8], ap + 8);
    async_copy16(&Bs[buf][lrow][lcol],     bp);
    async_copy16(&Bs[buf][lrow][lcol + 8], bp + 8);
  };

  const int nk = K >> 5;
  stage(0, 0);
  for (int kt = 0; kt < nk; ++kt) {
    async_wait_all();        // my tile-kt async loads have landed in LDS
    __syncthreads();         // everyone's have; everyone's reads of buf (kt+1)&1 are done
    const int cur = kt & 1;
    if (kt + 1 < nk) stage(1 - cur, kt + 1);   // overlap next-tile DMA with WMMA below
    if (kt + 4 < nk) {       // L2 prefetch (weights are L2-resident after first pass)
      __builtin_prefetch(aRow + (kt + 4) * 32, 0, 1);
      __builtin_prefetch(bRow + (kt + 4) * 32, 0, 1);
    }
    v16bf af[2], bfv[4];
#pragma unroll
    for (int mt = 0; mt < 2; ++mt) {
      const int row = wm * 32 + mt * 16 + l16;
      af[mt] = frag2(&As[cur][row][lhi * 16], &As[cur][row][lhi * 16 + 8]);
    }
#pragma unroll
    for (int nt = 0; nt < 4; ++nt) {
      const int col = wn * 64 + nt * 16 + l16;
      bfv[nt] = frag2(&Bs[cur][col][lhi * 16], &Bs[cur][col][lhi * 16 + 8]);
    }
#pragma unroll
    for (int mt = 0; mt < 2; ++mt)
#pragma unroll
      for (int nt = 0; nt < 4; ++nt)
        acc[mt][nt] = __builtin_amdgcn_wmma_f32_16x16x32_bf16(
            false, af[mt], false, bfv[nt], (short)0, acc[mt][nt], false, false);
  }

#pragma unroll
  for (int mt = 0; mt < 2; ++mt)
#pragma unroll
    for (int nt = 0; nt < 4; ++nt) {
      const int n = nBlk + wn * 64 + nt * 16 + l16;
      const float sc = scale[n];
#pragma unroll
      for (int r = 0; r < 8; ++r) {
        const int m = mBlk + wm * 32 + mt * 16 + r + 8 * lhi;  // C layout ISA 7.12.2
        const size_t idx = (size_t)m * N + n;
        const float v = acc[mt][nt][r] * sc;
        out[idx] = RESID ? (resid[idx] + v) : v;
      }
    }
}

// ---------------- RoPE + repack [s, nh*128]f32 -> [nh][s][128]bf16 ----------------
__global__ void rope_pack_k(const float* __restrict__ in, const float* __restrict__ cs,
                            const float* __restrict__ sn, bh* __restrict__ out,
                            int nh, int do_rope) {
  const int total = S_LEN * nh * DHEAD;
  const int idx = blockIdx.x * blockDim.x + threadIdx.x;
  if (idx >= total) return;
  const int cols = nh * DHEAD;
  const int s = idx / cols, c = idx % cols;
  const int head = c >> 7, d = c & 127;
  float v = in[idx];
  if (do_rope) {
    const float rh = (d < 64) ? -in[idx + 64] : in[idx - 64];
    v = v * cs[s * DHEAD + d] + rh * sn[s * DHEAD + d];
  }
  out[((size_t)head * S_LEN + s) * DHEAD + d] = (bh)v;
}

// ---------------- flash attention (GQA, full softmax), WMMA for QK^T and PV -------
// grid = (S/64, 20), block = 128 (4 waves); wave w owns q rows [64*bx + 16w, +16)
__global__ __launch_bounds__(128)
void attn_k(const bh* __restrict__ qh, const bh* __restrict__ kh,
            const bh* __restrict__ vh, bh* __restrict__ o) {
  __shared__ bh Qs[64][136];
  __shared__ bh Ks[32][136];
  __shared__ bh Vt[128][40];     // V transposed: Vt[d][kv]
  __shared__ bh Ps[4][16][32];   // per-wave P staging (C-layout -> A-layout)
  const int head = blockIdx.y;
  const int kvh  = head >> 2;    // n_rep = 4, repeat_interleave
  const int qt   = blockIdx.x;
  const int tid  = threadIdx.x;
  const int wave = tid >> 5, lane = tid & 31;
  const int l16 = lane & 15, lhi = lane >> 4;

  const bh* qp = qh + ((size_t)head * S_LEN + qt * 64) * DHEAD;
  for (int i = tid; i < 64 * 16; i += 128) {       // async Q tile load
    const int row = i >> 4, c8 = (i & 15) * 8;
    async_copy16(&Qs[row][c8], &qp[row * DHEAD + c8]);
  }

  float m[8], l[8];
  v8f oacc[8];
#pragma unroll
  for (int r = 0; r < 8; ++r) { m[r] = -1e30f; l[r] = 0.f; }
#pragma unroll
  for (int f = 0; f < 8; ++f)
#pragma unroll
    for (int i = 0; i < 8; ++i) oacc[f][i] = 0.f;

  const float scl = 0.08838834764831843f;  // 1/sqrt(128)
  for (int jt = 0; jt < S_LEN / 32; ++jt) {
    __syncthreads();
    const bh* kp = kh + ((size_t)kvh * S_LEN + jt * 32) * DHEAD;
    for (int i = tid; i < 32 * 16; i += 128) {     // async K tile load
      const int row = i >> 4, c8 = (i & 15) * 8;
      async_copy16(&Ks[row][c8], &kp[row * DHEAD + c8]);
    }
    const bh* vp = vh + ((size_t)kvh * S_LEN + jt * 32) * DHEAD;
    for (int i = tid; i < 32 * 128; i += 128) {    // transposing store, stays scalar
      const int kv = i >> 7, d = i & 127;
      Vt[d][kv] = vp[kv * DHEAD + d];
    }
    async_wait_all();
    __syncthreads();

    // S = Q (16x128) * K^T (128x32): two 16x16 C frags, K-loop of 4x32
    v8f s0, s1;
#pragma unroll
    for (int i = 0; i < 8; ++i) { s0[i] = 0.f; s1[i] = 0.f; }
    const int qrow = wave * 16 + l16;
#pragma unroll
    for (int kc = 0; kc < 4; ++kc) {
      v16bf af = frag2(&Qs[qrow][kc * 32 + lhi * 16], &Qs[qrow][kc * 32 + lhi * 16 + 8]);
      v16bf b0 = frag2(&Ks[l16][kc * 32 + lhi * 16],      &Ks[l16][kc * 32 + lhi * 16 + 8]);
      v16bf b1 = frag2(&Ks[16 + l16][kc * 32 + lhi * 16], &Ks[16 + l16][kc * 32 + lhi * 16 + 8]);
      s0 = __builtin_amdgcn_wmma_f32_16x16x32_bf16(false, af, false, b0, (short)0, s0, false, false);
      s1 = __builtin_amdgcn_wmma_f32_16x16x32_bf16(false, af, false, b1, (short)0, s1, false, false);
    }

    // online softmax: C frag row = r + 8*lhi, col = l16 (+16 for s1)
    float tmax[8], tsum[8], alpha[8];
#pragma unroll
    for (int r = 0; r < 8; ++r) {
      s0[r] *= scl; s1[r] *= scl;
      tmax[r] = fmaxf(s0[r], s1[r]);
    }
#pragma unroll
    for (int off = 1; off < 16; off <<= 1)
#pragma unroll
      for (int r = 0; r < 8; ++r)
        tmax[r] = fmaxf(tmax[r], __shfl_xor(tmax[r], off, 32));
#pragma unroll
    for (int r = 0; r < 8; ++r) {
      const float mn = fmaxf(m[r], tmax[r]);
      alpha[r] = __expf(m[r] - mn);
      const float p0 = __expf(s0[r] - mn);
      const float p1 = __expf(s1[r] - mn);
      const int rowl = r + 8 * lhi;
      Ps[wave][rowl][l16]      = (bh)p0;
      Ps[wave][rowl][16 + l16] = (bh)p1;
      tsum[r] = p0 + p1;
      m[r] = mn;
    }
#pragma unroll
    for (int off = 1; off < 16; off <<= 1)
#pragma unroll
      for (int r = 0; r < 8; ++r)
        tsum[r] += __shfl_xor(tsum[r], off, 32);
#pragma unroll
    for (int r = 0; r < 8; ++r) l[r] = l[r] * alpha[r] + tsum[r];
#pragma unroll
    for (int f = 0; f < 8; ++f)
#pragma unroll
      for (int r = 0; r < 8; ++r) oacc[f][r] *= alpha[r];

    // P (16x32, A-frag from LDS) x V (32x128): 8 N-tiles
    v16bf pf = frag2(&Ps[wave][l16][lhi * 16], &Ps[wave][l16][lhi * 16 + 8]);
#pragma unroll
    for (int f = 0; f < 8; ++f) {
      v16bf bv = frag2(&Vt[f * 16 + l16][lhi * 16], &Vt[f * 16 + l16][lhi * 16 + 8]);
      oacc[f] = __builtin_amdgcn_wmma_f32_16x16x32_bf16(false, pf, false, bv, (short)0, oacc[f], false, false);
    }
  }

#pragma unroll
  for (int f = 0; f < 8; ++f)
#pragma unroll
    for (int r = 0; r < 8; ++r) {
      const int srow = qt * 64 + wave * 16 + r + 8 * lhi;
      const int col  = head * DHEAD + f * 16 + l16;
      o[(size_t)srow * HDIM + col] = (bh)(oacc[f][r] / l[r]);
    }
}

// ---------------- relu^2 GLU: out = bf16(relu(gate)^2 * up) ----------------
__global__ void glu_k(const float* __restrict__ gate, const float* __restrict__ up,
                      bh* __restrict__ out, int total) {
  const int i = blockIdx.x * blockDim.x + threadIdx.x;
  if (i >= total) return;
  const float g = fmaxf(gate[i], 0.f);
  out[i] = (bh)(g * g * up[i]);
}

// =========================== launcher ===========================
extern "C" void kernel_launch(void* const* d_in, const int* in_sizes, int n_in,
                              void* d_out, int out_size, void* d_ws, size_t ws_size,
                              hipStream_t stream) {
  (void)in_sizes; (void)n_in; (void)out_size; (void)ws_size;
  const float* x    = (const float*)d_in[0];
  const float* cosr = (const float*)d_in[1];
  const float* sinr = (const float*)d_in[2];
  const float* q_w  = (const float*)d_in[3];
  const float* k_w  = (const float*)d_in[4];
  const float* v_w  = (const float*)d_in[5];
  const float* o_w  = (const float*)d_in[6];
  const float* g_w  = (const float*)d_in[7];
  const float* u_w  = (const float*)d_in[8];
  const float* dn_w = (const float*)d_in[9];
  const float* ln1  = (const float*)d_in[10];
  const float* ln2  = (const float*)d_in[11];
  float* out = (float*)d_out;

  size_t off = 0;
  auto alloc = [&](size_t bytes) -> void* {
    void* p = (char*)d_ws + off;
    off += (bytes + 255) & ~(size_t)255;
    return p;
  };
  // ternary weights (bf16) — 139 MB total, fits in the 192 MB L2
  bh* tq = (bh*)alloc((size_t)HDIM  * HDIM  * 2);
  bh* tk = (bh*)alloc((size_t)KVDIM * HDIM  * 2);
  bh* tv = (bh*)alloc((size_t)KVDIM * HDIM  * 2);
  bh* to_ = (bh*)alloc((size_t)HDIM * HDIM  * 2);
  bh* tg = (bh*)alloc((size_t)INTER * HDIM  * 2);
  bh* tu = (bh*)alloc((size_t)INTER * HDIM  * 2);
  bh* td = (bh*)alloc((size_t)HDIM  * INTER * 2);
  float* sq = (float*)alloc(HDIM  * 4);
  float* sk = (float*)alloc(KVDIM * 4);
  float* sv = (float*)alloc(KVDIM * 4);
  float* so = (float*)alloc(HDIM  * 4);
  float* sg = (float*)alloc(INTER * 4);
  float* su = (float*)alloc(INTER * 4);
  float* sd = (float*)alloc(HDIM  * 4);
  // activations
  bh*    h1  = (bh*)   alloc((size_t)S_LEN * HDIM  * 2);
  float* qf  = (float*)alloc((size_t)S_LEN * HDIM  * 4);
  float* kf  = (float*)alloc((size_t)S_LEN * KVDIM * 4);
  float* vf  = (float*)alloc((size_t)S_LEN * KVDIM * 4);
  bh*    qhb = (bh*)   alloc((size_t)S_LEN * HDIM  * 2);
  bh*    khb = (bh*)   alloc((size_t)S_LEN * KVDIM * 2);
  bh*    vhb = (bh*)   alloc((size_t)S_LEN * KVDIM * 2);
  bh*    ob  = (bh*)   alloc((size_t)S_LEN * HDIM  * 2);
  float* x2  = (float*)alloc((size_t)S_LEN * HDIM  * 4);
  bh*    h2  = (bh*)   alloc((size_t)S_LEN * HDIM  * 2);
  float* gf  = (float*)alloc((size_t)S_LEN * INTER * 4);
  float* uf  = (float*)alloc((size_t)S_LEN * INTER * 4);
  bh*    gub = (bh*)   alloc((size_t)S_LEN * INTER * 2);

  // 1) ternarize all weights
  ternarize_k<<<HDIM,  256, 0, stream>>>(q_w,  tq,  sq, HDIM);
  ternarize_k<<<KVDIM, 256, 0, stream>>>(k_w,  tk,  sk, HDIM);
  ternarize_k<<<KVDIM, 256, 0, stream>>>(v_w,  tv,  sv, HDIM);
  ternarize_k<<<HDIM,  256, 0, stream>>>(o_w,  to_, so, HDIM);
  ternarize_k<<<INTER, 256, 0, stream>>>(g_w,  tg,  sg, HDIM);
  ternarize_k<<<INTER, 256, 0, stream>>>(u_w,  tu,  su, HDIM);
  ternarize_k<<<HDIM,  256, 0, stream>>>(dn_w, td,  sd, INTER);

  // 2) attention half
  rmsnorm_k<<<S_LEN, 256, 0, stream>>>(x, ln1, h1);
  gemm_tern<false><<<dim3(HDIM  / 128, S_LEN / 128), 256, 0, stream>>>(h1, tq, sq, nullptr, qf, HDIM,  HDIM);
  gemm_tern<false><<<dim3(KVDIM / 128, S_LEN / 128), 256, 0, stream>>>(h1, tk, sk, nullptr, kf, KVDIM, HDIM);
  gemm_tern<false><<<dim3(KVDIM / 128, S_LEN / 128), 256, 0, stream>>>(h1, tv, sv, nullptr, vf, KVDIM, HDIM);
  rope_pack_k<<<(S_LEN * HDIM  + 255) / 256, 256, 0, stream>>>(qf, cosr, sinr, qhb, NHEADS, 1);
  rope_pack_k<<<(S_LEN * KVDIM + 255) / 256, 256, 0, stream>>>(kf, cosr, sinr, khb, NKV, 1);
  rope_pack_k<<<(S_LEN * KVDIM + 255) / 256, 256, 0, stream>>>(vf, cosr, sinr, vhb, NKV, 0);
  attn_k<<<dim3(S_LEN / 64, NHEADS), 128, 0, stream>>>(qhb, khb, vhb, ob);
  gemm_tern<true><<<dim3(HDIM / 128, S_LEN / 128), 256, 0, stream>>>(ob, to_, so, x, x2, HDIM, HDIM);

  // 3) MLP half
  rmsnorm_k<<<S_LEN, 256, 0, stream>>>(x2, ln2, h2);
  gemm_tern<false><<<dim3(INTER / 128, S_LEN / 128), 256, 0, stream>>>(h2, tg, sg, nullptr, gf, INTER, HDIM);
  gemm_tern<false><<<dim3(INTER / 128, S_LEN / 128), 256, 0, stream>>>(h2, tu, su, nullptr, uf, INTER, HDIM);
  glu_k<<<(S_LEN * INTER + 255) / 256, 256, 0, stream>>>(gf, uf, gub, S_LEN * INTER);
  gemm_tern<true><<<dim3(HDIM / 128, S_LEN / 128), 256, 0, stream>>>(gub, td, sd, x2, out, HDIM, INTER);
}
